// MultiheadSelfAttention_53420803228234
// MI455X (gfx1250) — matrix operand order repaired and checked by reading, not compiled, not run on hardware
//
#include <hip/hip_runtime.h>

typedef __bf16 bf16_t;
typedef __attribute__((ext_vector_type(16))) __bf16 v16bf;
typedef __attribute__((ext_vector_type(8)))  float  v8f;
typedef unsigned int u32x4 __attribute__((ext_vector_type(4)));
typedef int          i32x4 __attribute__((ext_vector_type(4)));
typedef int          i32x8 __attribute__((ext_vector_type(8)));

#define B_   2
#define T_   2048
#define DM_  1024
#define DH_  128
#define NH_  8

// LDS strides (bf16 elements; every row start 16B-aligned; strides match the
// TDM pad encodings used below)
#define XS_LD 72    // 64 data + 8 pad   (32 DW + 4 DW)
#define WS_LD 72
#define CS_LD 132   // fp32 stage, 128 + 4
#define KS_LD 136   // 128 data + 8 pad  (64 DW + 4 DW)
#define VT_LD 40    // 32 data + 8 pad   (16 DW + 4 DW)
#define PS_LD 40

#if __has_builtin(__builtin_amdgcn_tensor_load_to_lds)
#define HAVE_TDM 1
#else
#define HAVE_TDM 0
#endif

__device__ __forceinline__ v8f wmma_bf16(v16bf a, v16bf b, v8f c) {
  return __builtin_amdgcn_wmma_f32_16x16x32_bf16(false, a, false, b, (short)0, c,
                                                 false, false);
}

// A fragment (16x32 MxK): lanes 0-15 row M=lane hold K={0..7,16..23},
// lanes 16-31 row M=lane-16 hold K={8..15,24..31}. Buffer row-major [M][K].
__device__ __forceinline__ v16bf frag_a(const bf16_t* __restrict__ p0, int ld, int lane) {
  const int m  = lane & 15;
  const int ko = (lane & 16) ? 8 : 0;
  const bf16_t* p = p0 + m * ld + ko;
  v16bf f;
#pragma unroll
  for (int i = 0; i < 8; ++i) f[i] = p[i];
#pragma unroll
  for (int i = 0; i < 8; ++i) f[8 + i] = p[16 + i];
  return f;
}

// B fragment (32x16 KxN): lane holds column N=lane&15; lanes 0-15 K={0..7,16..23},
// lanes 16-31 K={8..15,24..31}. Buffer N-major [N][K].
__device__ __forceinline__ v16bf frag_b(const bf16_t* __restrict__ p0, int ld, int lane) {
  const int n  = lane & 15;
  const int ko = (lane & 16) ? 8 : 0;
  const bf16_t* p = p0 + n * ld + ko;
  v16bf f;
#pragma unroll
  for (int i = 0; i < 8; ++i) f[i] = p[i];
#pragma unroll
  for (int i = 0; i < 8; ++i) f[8 + i] = p[16 + i];
  return f;
}

__device__ __forceinline__ unsigned lds_addr_of(const void* p) {
  // flat LDS address: addr[31:0] is the LDS byte offset (aperture rule)
  return (unsigned)(unsigned long long)p;
}

#if HAVE_TDM
// Wait helper with the required constant immediates.
__device__ __forceinline__ void tdm_wait(bool more) {
  if (more) __builtin_amdgcn_s_wait_tensorcnt(2);   // newest pair may be in flight
  else      __builtin_amdgcn_s_wait_tensorcnt(0);   // drain everything
}

// 2D bf16 tile DMA: global tile [tile_d1 rows x tile_d0 elems], row stride
// d0_stride elems -> LDS at lds_addr with per-row padding.
__device__ __forceinline__ void tdm_load_2d_bf16(
    unsigned lds_addr, const void* gptr,
    unsigned tile_d0, unsigned tile_d1,
    unsigned d0_stride,
    unsigned pad_interval_code, unsigned pad_amount_code)
{
  unsigned long long ga = (unsigned long long)gptr;
  u32x4 g0;
  g0[0] = 1u;                                                // count=1, user D#
  g0[1] = lds_addr;
  g0[2] = (unsigned)ga;
  g0[3] = (unsigned)((ga >> 32) & 0x01FFFFFFu) | (2u << 30); // addr[56:32], type=2
  i32x8 g1;
  g1[0] = (int)((1u << 16) |                                 // data_size=1 (2B)
                (1u << 20) |                                 // pad_enable
                (pad_interval_code << 22) | (pad_amount_code << 25));
  g1[1] = (int)((tile_d0 & 0xFFFFu) << 16);                  // tensor_dim0 = tile_d0
  g1[2] = (int)(((tile_d0 >> 16) & 0xFFFFu) |
                ((tile_d1 & 0xFFFFu) << 16));                // tensor_dim1 = tile_d1
  g1[3] = (int)(((tile_d1 >> 16) & 0xFFFFu) |
                ((tile_d0 & 0xFFFFu) << 16));                // tile_dim0
  g1[4] = (int)(tile_d1 & 0xFFFFu);                          // tile_dim1 (dim2=0)
  g1[5] = (int)d0_stride;                                    // dim0_stride[31:0]
  g1[6] = 0;
  g1[7] = 0;
  i32x4 z4 = {0, 0, 0, 0};
#if defined(__clang_major__) && (__clang_major__ >= 23)
  i32x8 z8 = {0, 0, 0, 0, 0, 0, 0, 0};
  __builtin_amdgcn_tensor_load_to_lds(g0, g1, z4, z4, z8, 0);
#else
  __builtin_amdgcn_tensor_load_to_lds(g0, g1, z4, z4, 0);
#endif
}
#endif

// ---------------------------------------------------------------------------
// Kernel 0a: bulk fp32 -> bf16 (x)
// ---------------------------------------------------------------------------
__global__ __launch_bounds__(256) void cvt_f32_to_bf16(const float* __restrict__ in,
                                                       bf16_t* __restrict__ out,
                                                       int n4) {
  int i = blockIdx.x * 256 + threadIdx.x;
  if (i >= n4) return;
  float4 v = ((const float4*)in)[i];
  bf16_t h[4] = {(bf16_t)v.x, (bf16_t)v.y, (bf16_t)v.z, (bf16_t)v.w};
  uint2 u;
  __builtin_memcpy(&u, h, 8);
  ((uint2*)out)[i] = u;
}

// ---------------------------------------------------------------------------
// Kernel 0b: fp32 [S][R][C] -> bf16 [S][C][R]  (one-shot weight transpose)
// ---------------------------------------------------------------------------
__global__ __launch_bounds__(256) void cvt_t_f32_to_bf16(const float* __restrict__ in,
                                                         bf16_t* __restrict__ outT,
                                                         int R, int C, int total) {
  int idx = blockIdx.x * 256 + threadIdx.x;
  if (idx >= total) return;
  int rc = R * C;
  int s = idx / rc;
  int rem = idx - s * rc;
  int r = rem / C, c = rem - (rem / C) * C;
  outT[(size_t)s * rc + (size_t)c * R + r] = (bf16_t)in[idx];
}

// ---------------------------------------------------------------------------
// Kernel 1: QKV projection + fused RoPE + muP scale.  Double-buffered TDM
// tiles; 8 waves, wave w owns output cols [16w,16w+16); K-chunk 64.
// V is written pre-transposed [bh][dh][T] for the attention kernel.
// ---------------------------------------------------------------------------
__global__ __launch_bounds__(256) void qkv_rope_kernel(
    const bf16_t* __restrict__ xb,                 // [B*T, DM] bf16
    const bf16_t* __restrict__ wT,                 // [3][NH][DH][DM] bf16 (q,k,v)
    bf16_t* __restrict__ qout,                     // [B*NH, T, DH]
    bf16_t* __restrict__ kout,                     // [B*NH, T, DH]
    bf16_t* __restrict__ voutT)                    // [B*NH, DH, T]
{
  __shared__ bf16_t Xs[2][64 * XS_LD];
  __shared__ bf16_t Ws[2][128 * WS_LD];
  __shared__ float  Cs[64 * CS_LD];

  const int tid   = threadIdx.x;
  const int lane  = tid & 31;
  const int wave  = tid >> 5;
  const int t0    = blockIdx.x * 64;
  const int which = blockIdx.y;                    // 0=q 1=k 2=v
  const int bh    = blockIdx.z;
  const int b     = bh / NH_;
  const int h     = bh % NH_;

  const bf16_t* w  = wT + ((size_t)which * NH_ + h) * DH_ * DM_;   // [DH][DM]
  const bf16_t* xp = xb + ((size_t)b * T_ + t0) * DM_;

  v8f acc[4];
#pragma unroll
  for (int rg = 0; rg < 4; ++rg)
#pragma unroll
    for (int i = 0; i < 8; ++i) acc[rg][i] = 0.0f;

  const int NIT = DM_ / 64;
#if HAVE_TDM
  if (wave == 0) {
    tdm_load_2d_bf16(lds_addr_of(&Xs[0][0]), xp, 64, 64, DM_, 4, 3);
    tdm_load_2d_bf16(lds_addr_of(&Ws[0][0]), w,  64, 128, DM_, 4, 3);
  }
#endif
  for (int it = 0; it < NIT; ++it) {
    const int cur = it & 1;
#if HAVE_TDM
    if (wave == 0) {
      if (it + 1 < NIT) {
        const int k1 = (it + 1) * 64;
        tdm_load_2d_bf16(lds_addr_of(&Xs[cur ^ 1][0]), xp + k1, 64, 64, DM_, 4, 3);
        tdm_load_2d_bf16(lds_addr_of(&Ws[cur ^ 1][0]), w + k1,  64, 128, DM_, 4, 3);
      }
      tdm_wait(it + 1 < NIT);
    }
#else
    {
      const int k0 = it * 64;
#pragma unroll
      for (int i = 0; i < 2; ++i) {
        int chunk = tid + i * 256;                 // 0..511, 8 chunks/row
        int r = chunk >> 3, c8 = (chunk & 7) * 8;
        *(uint4*)&Xs[cur][r * XS_LD + c8] =
            *(const uint4*)&xp[(size_t)r * DM_ + k0 + c8];
      }
#pragma unroll
      for (int i = 0; i < 4; ++i) {
        int chunk = tid + i * 256;                 // 0..1023, 8 chunks/n-row
        int n = chunk >> 3, k8 = (chunk & 7) * 8;
        *(uint4*)&Ws[cur][n * WS_LD + k8] =
            *(const uint4*)&w[(size_t)n * DM_ + k0 + k8];
      }
    }
#endif
    __syncthreads();
    v16bf bf0 = frag_b(&Ws[cur][(wave * 16) * WS_LD], WS_LD, lane);
    v16bf bf1 = frag_b(&Ws[cur][(wave * 16) * WS_LD + 32], WS_LD, lane);
#pragma unroll
    for (int rg = 0; rg < 4; ++rg) {
      v16bf af0 = frag_a(&Xs[cur][(rg * 16) * XS_LD], XS_LD, lane);
      v16bf af1 = frag_a(&Xs[cur][(rg * 16) * XS_LD + 32], XS_LD, lane);
      acc[rg] = wmma_bf16(af0, bf0, acc[rg]);
      acc[rg] = wmma_bf16(af1, bf1, acc[rg]);
    }
    __syncthreads();
  }

  // Stage fp32 C (64x128) to LDS for cross-tile RoPE pairing.
  const int halfl = (lane >> 4) & 1;
  const int cn    = wave * 16 + (lane & 15);
#pragma unroll
  for (int rg = 0; rg < 4; ++rg)
#pragma unroll
    for (int r = 0; r < 8; ++r)
      Cs[(rg * 16 + r + halfl * 8) * CS_LD + cn] = acc[rg][r];
  __syncthreads();

  if (which < 2) {
    const float scale = 0.08838834764831845f;      // 128^-0.5 (applied to q AND k)
    bf16_t* outp = (which == 0 ? qout : kout) + ((size_t)bh * T_ + t0) * DH_;
#pragma unroll
    for (int i = 0; i < 8; ++i) {
      int idx = tid + i * 256;                     // 64 rows * 32 pairs
      int row = idx >> 5, c = idx & 31;
      float ang = (float)(t0 + row) * powf(10000.0f, -(float)c * (1.0f / 32.0f));
      float sn, cs2;
      sincosf(ang, &sn, &cs2);
      float e = Cs[row * CS_LD + c];
      float o = Cs[row * CS_LD + c + 32];
      outp[row * DH_ + c]      = (bf16_t)((e * cs2 - o * sn) * scale);
      outp[row * DH_ + c + 32] = (bf16_t)((e * sn + o * cs2) * scale);
    }
#pragma unroll
    for (int i = 0; i < 16; ++i) {
      int idx = tid + i * 256;                     // 64 rows * 64 pass-through
      int row = idx >> 6, c = 64 + (idx & 63);
      outp[row * DH_ + c] = (bf16_t)(Cs[row * CS_LD + c] * scale);
    }
  } else {
    // V: write transposed [bh][dh][T]; tid-major along T for coalescing
#pragma unroll
    for (int i = 0; i < 32; ++i) {
      int idx = tid + i * 256;                     // 0..8191
      int row = idx & 63, c = idx >> 6;
      voutT[((size_t)bh * DH_ + c) * T_ + t0 + row] = (bf16_t)Cs[row * CS_LD + c];
    }
  }
}

// ---------------------------------------------------------------------------
// Kernel 2: causal flash attention.  4 waves; K tile (32x128) and V^T tile
// (128x32) double-buffered via TDM, overlapping DMA(kt+1) with WMMA(kt).
// ---------------------------------------------------------------------------
__global__ __launch_bounds__(128) void flash_attn_kernel(
    const bf16_t* __restrict__ Q,                  // [B*NH, T, DH]
    const bf16_t* __restrict__ K,                  // [B*NH, T, DH]
    const bf16_t* __restrict__ Vt_g,               // [B*NH, DH, T]
    bf16_t* __restrict__ O)                        // [B, T, NH*DH]
{
  __shared__ bf16_t Ks[2][32 * KS_LD];
  __shared__ bf16_t Vt[2][128 * VT_LD];
  __shared__ bf16_t Ps[64 * PS_LD];

  const int tid   = threadIdx.x;
  const int lane  = tid & 31;
  const int wave  = tid >> 5;
  const int t0    = blockIdx.x * 64;
  const int h     = blockIdx.y;
  const int b     = blockIdx.z;
  const int bh    = b * NH_ + h;
  const int halfl = (lane >> 4) & 1;
  const int jc    = lane & 15;

  const bf16_t* Qp  = Q + ((size_t)bh * T_ + t0 + wave * 16) * DH_;
  const bf16_t* Kp  = K + (size_t)bh * T_ * DH_;
  const bf16_t* Vtp = Vt_g + (size_t)bh * DH_ * T_;

  v16bf qf[4];
#pragma unroll
  for (int kk = 0; kk < 4; ++kk) qf[kk] = frag_a(Qp + kk * 32, DH_, lane);

  v8f   o_acc[8];
  float m_r[8], l_r[8];
#pragma unroll
  for (int r = 0; r < 8; ++r) {
    m_r[r] = -3.0e38f;
    l_r[r] = 0.0f;
#pragma unroll
    for (int n = 0; n < 8; ++n) o_acc[n][r] = 0.0f;
  }

  const int nkt = (t0 + 64) / 32;                  // causal: keys [0, t0+64)
#if HAVE_TDM
  if (wave == 0) {
    tdm_load_2d_bf16(lds_addr_of(&Ks[0][0]), Kp,  DH_, 32, DH_, 5, 3);
    tdm_load_2d_bf16(lds_addr_of(&Vt[0][0]), Vtp, 32, DH_, T_, 3, 3);
  }
#endif
  for (int kt = 0; kt < nkt; ++kt) {
    const int kbase = kt * 32;
    const int cur = kt & 1;
#if HAVE_TDM
    if (wave == 0) {
      if (kt + 1 < nkt) {
        const int kb1 = kbase + 32;
        tdm_load_2d_bf16(lds_addr_of(&Ks[cur ^ 1][0]), Kp + (size_t)kb1 * DH_,
                         DH_, 32, DH_, 5, 3);
        tdm_load_2d_bf16(lds_addr_of(&Vt[cur ^ 1][0]), Vtp + kb1,
                         32, DH_, T_, 3, 3);
      }
      tdm_wait(kt + 1 < nkt);
    }
#else
#pragma unroll
    for (int i = 0; i < 4; ++i) {
      int chunk = tid + i * 128;                   // 0..511
      int r = chunk >> 4, c8 = (chunk & 15) * 8;
      *(uint4*)&Ks[cur][r * KS_LD + c8] =
          *(const uint4*)&Kp[(size_t)(kbase + r) * DH_ + c8];
      int c = chunk >> 2, k8 = (chunk & 3) * 8;
      *(uint4*)&Vt[cur][c * VT_LD + k8] =
          *(const uint4*)&Vtp[(size_t)c * T_ + kbase + k8];
    }
    if (kt + 1 < nkt)
      __builtin_prefetch(Kp + (size_t)(kbase + 32) * DH_ + tid * 64, 0, 1);
#endif
    __syncthreads();

    // S = Q K^T : two 16x16 C tiles
    v8f s0, s1;
#pragma unroll
    for (int r = 0; r < 8; ++r) { s0[r] = 0.0f; s1[r] = 0.0f; }
#pragma unroll
    for (int kk = 0; kk < 4; ++kk) {
      v16bf b0 = frag_b(&Ks[cur][0 * KS_LD + kk * 32], KS_LD, lane);
      v16bf b1 = frag_b(&Ks[cur][16 * KS_LD + kk * 32], KS_LD, lane);
      s0 = wmma_bf16(qf[kk], b0, s0);
      s1 = wmma_bf16(qf[kk], b1, s1);
    }

    // mask + streaming softmax (row = vr + 8*halfl within wave's 16 rows)
#pragma unroll
    for (int r = 0; r < 8; ++r) {
      const int il  = r + halfl * 8;
      const int i_g = t0 + wave * 16 + il;
      float v0 = (kbase + jc      <= i_g) ? s0[r] : -1.0e30f;
      float v1 = (kbase + 16 + jc <= i_g) ? s1[r] : -1.0e30f;
      float mt = fmaxf(v0, v1);
#pragma unroll
      for (int off = 8; off; off >>= 1) mt = fmaxf(mt, __shfl_xor(mt, off, 32));
      float mn = fmaxf(m_r[r], mt);
      float alpha = __expf(m_r[r] - mn);
      m_r[r] = mn;
      float p0 = __expf(v0 - mn);
      float p1 = __expf(v1 - mn);
      float lt = p0 + p1;
#pragma unroll
      for (int off = 8; off; off >>= 1) lt += __shfl_xor(lt, off, 32);
      l_r[r] = l_r[r] * alpha + lt;
      Ps[(wave * 16 + il) * PS_LD + jc]      = (bf16_t)p0;
      Ps[(wave * 16 + il) * PS_LD + 16 + jc] = (bf16_t)p1;
#pragma unroll
      for (int n = 0; n < 8; ++n) o_acc[n][r] *= alpha;
    }

    // O += P V
    v16bf pf = frag_a(&Ps[(wave * 16) * PS_LD], PS_LD, lane);
#pragma unroll
    for (int n = 0; n < 8; ++n) {
      v16bf vf = frag_b(&Vt[cur][(n * 16) * VT_LD], VT_LD, lane);
      o_acc[n] = wmma_bf16(pf, vf, o_acc[n]);
    }
    __syncthreads();
  }

  // normalize, write O bf16 [B, T, NH*DH]
#pragma unroll
  for (int r = 0; r < 8; ++r) {
    float inv = 1.0f / l_r[r];
    int t = t0 + wave * 16 + r + halfl * 8;
    bf16_t* row = O + ((size_t)b * T_ + t) * DM_ + h * DH_;
#pragma unroll
    for (int n = 0; n < 8; ++n)
      row[n * 16 + jc] = (bf16_t)(o_acc[n][r] * inv);
  }
}

// ---------------------------------------------------------------------------
// Kernel 3: output projection  out[4096,1024] = O_bf16 @ wo (woT pre-staged)
// ---------------------------------------------------------------------------
__global__ __launch_bounds__(256) void out_proj_kernel(
    const bf16_t* __restrict__ Oin,                // [B*T, DM] bf16
    const bf16_t* __restrict__ woT,                // [DM(o)][DM(d)] bf16
    float* __restrict__ out)
{
  __shared__ bf16_t Xs[2][64 * XS_LD];
  __shared__ bf16_t Ws[2][128 * WS_LD];

  const int tid  = threadIdx.x;
  const int lane = tid & 31;
  const int wave = tid >> 5;
  const int row0 = blockIdx.x * 64;
  const int col0 = blockIdx.y * 128;

  const bf16_t* xp = Oin + (size_t)row0 * DM_;
  const bf16_t* wp = woT + (size_t)col0 * DM_;

  v8f acc[4];
#pragma unroll
  for (int rg = 0; rg < 4; ++rg)
#pragma unroll
    for (int i = 0; i < 8; ++i) acc[rg][i] = 0.0f;

  const int NIT = DM_ / 64;
#if HAVE_TDM
  if (wave == 0) {
    tdm_load_2d_bf16(lds_addr_of(&Xs[0][0]), xp, 64, 64, DM_, 4, 3);
    tdm_load_2d_bf16(lds_addr_of(&Ws[0][0]), wp, 64, 128, DM_, 4, 3);
  }
#endif
  for (int it = 0; it < NIT; ++it) {
    const int cur = it & 1;
#if HAVE_TDM
    if (wave == 0) {
      if (it + 1 < NIT) {
        const int k1 = (it + 1) * 64;
        tdm_load_2d_bf16(lds_addr_of(&Xs[cur ^ 1][0]), xp + k1, 64, 64, DM_, 4, 3);
        tdm_load_2d_bf16(lds_addr_of(&Ws[cur ^ 1][0]), wp + k1, 64, 128, DM_, 4, 3);
      }
      tdm_wait(it + 1 < NIT);
    }
#else
    {
      const int k0 = it * 64;
#pragma unroll
      for (int i = 0; i < 2; ++i) {
        int chunk = tid + i * 256;
        int r = chunk >> 3, c8 = (chunk & 7) * 8;
        *(uint4*)&Xs[cur][r * XS_LD + c8] =
            *(const uint4*)&xp[(size_t)r * DM_ + k0 + c8];
      }
#pragma unroll
      for (int i = 0; i < 4; ++i) {
        int chunk = tid + i * 256;
        int n = chunk >> 3, k8 = (chunk & 7) * 8;
        *(uint4*)&Ws[cur][n * WS_LD + k8] =
            *(const uint4*)&wp[(size_t)n * DM_ + k0 + k8];
      }
    }
#endif
    __syncthreads();
    v16bf bf0 = frag_b(&Ws[cur][(wave * 16) * WS_LD], WS_LD, lane);
    v16bf bf1 = frag_b(&Ws[cur][(wave * 16) * WS_LD + 32], WS_LD, lane);
#pragma unroll
    for (int rg = 0; rg < 4; ++rg) {
      v16bf af0 = frag_a(&Xs[cur][(rg * 16) * XS_LD], XS_LD, lane);
      v16bf af1 = frag_a(&Xs[cur][(rg * 16) * XS_LD + 32], XS_LD, lane);
      acc[rg] = wmma_bf16(af0, bf0, acc[rg]);
      acc[rg] = wmma_bf16(af1, bf1, acc[rg]);
    }
    __syncthreads();
  }

  const int halfl = (lane >> 4) & 1;
  const int cn    = col0 + wave * 16 + (lane & 15);
#pragma unroll
  for (int rg = 0; rg < 4; ++rg)
#pragma unroll
    for (int r = 0; r < 8; ++r)
      out[(size_t)(row0 + rg * 16 + r + halfl * 8) * DM_ + cn] = acc[rg][r];
}

extern "C" void kernel_launch(void* const* d_in, const int* in_sizes, int n_in,
                              void* d_out, int out_size, void* d_ws, size_t ws_size,
                              hipStream_t stream) {
  (void)in_sizes; (void)n_in; (void)out_size; (void)ws_size;
  const float* x  = (const float*)d_in[0];
  const float* wq = (const float*)d_in[1];
  const float* wk = (const float*)d_in[2];
  const float* wv = (const float*)d_in[3];
  const float* wo = (const float*)d_in[4];
  float* out = (float*)d_out;

  const size_t x_elems   = (size_t)B_ * T_ * DM_;         // 4 Mi
  const size_t w_elems   = (size_t)NH_ * DM_ * DH_;       // 1 Mi
  const size_t qkv_elems = (size_t)B_ * NH_ * T_ * DH_;   // 4 Mi

  char* ws = (char*)d_ws;
  bf16_t* Xb   = (bf16_t*)ws;  ws += x_elems * 2;
  bf16_t* WT   = (bf16_t*)ws;  ws += 3 * w_elems * 2;     // [3][NH][DH][DM]
  bf16_t* WoT  = (bf16_t*)ws;  ws += w_elems * 2;         // [DM][DM] transposed
  bf16_t* Qb   = (bf16_t*)ws;  ws += qkv_elems * 2;
  bf16_t* Kb   = (bf16_t*)ws;  ws += qkv_elems * 2;
  bf16_t* VbT  = (bf16_t*)ws;  ws += qkv_elems * 2;       // [B*NH][DH][T]
  bf16_t* Ob   = (bf16_t*)ws;

  cvt_f32_to_bf16<<<(int)(x_elems / 4 / 256), 256, 0, stream>>>(x, Xb, (int)(x_elems / 4));
  const int wtot = (int)w_elems;                           // 1 Mi elems each
  cvt_t_f32_to_bf16<<<(wtot + 255) / 256, 256, 0, stream>>>(wq, WT + 0 * w_elems, DM_, DH_, wtot);
  cvt_t_f32_to_bf16<<<(wtot + 255) / 256, 256, 0, stream>>>(wk, WT + 1 * w_elems, DM_, DH_, wtot);
  cvt_t_f32_to_bf16<<<(wtot + 255) / 256, 256, 0, stream>>>(wv, WT + 2 * w_elems, DM_, DH_, wtot);
  cvt_t_f32_to_bf16<<<(wtot + 255) / 256, 256, 0, stream>>>(wo, WoT, DM_, DM_, wtot);

  dim3 g1(T_ / 64, 3, B_ * NH_);
  qkv_rope_kernel<<<g1, 256, 0, stream>>>(Xb, WT, Qb, Kb, VbT);

  dim3 g2(T_ / 64, NH_, B_);
  flash_attn_kernel<<<g2, 128, 0, stream>>>(Qb, Kb, VbT, Ob);

  dim3 g3((B_ * T_) / 64, DM_ / 128);
  out_proj_kernel<<<g3, 256, 0, stream>>>(Ob, WoT, out);
}
